// GTEATransLayer_1331439862432
// MI455X (gfx1250) — compile-verified, matrix operand order checked by reading, not run on hardware
//
#include <hip/hip_runtime.h>
#include <hip/hip_bf16.h>

typedef __attribute__((ext_vector_type(16))) _Float16 v16h;
typedef __attribute__((ext_vector_type(8)))  float    v8f;

#define EE      65536      // edges
#define SS      8          // seq len
#define ROWS_ES (EE*SS)    // 524288 rows
#define NDST    4096
#define HH      128

union Frag { v16h v; _Float16 h[16]; unsigned u[8]; };

// ---- CDNA5 async copy: global -> LDS, 16B per lane, tracked by ASYNCcnt ----
__device__ __forceinline__ void async_copy16(unsigned lds_byte, const _Float16* gsrc) {
  asm volatile("global_load_async_to_lds_b128 %0, %1, off"
               :: "v"(lds_byte), "v"(gsrc)
               : "memory");
}
__device__ __forceinline__ void wait_async0() {
  asm volatile("s_wait_asynccnt 0x0" ::: "memory");
}

// ---- 16-lane (half-wave) sum reduction; xor masks never cross the 16-lane group (wave32) ----
__device__ __forceinline__ float hred16(float x) {
  x += __shfl_xor(x, 1);
  x += __shfl_xor(x, 2);
  x += __shfl_xor(x, 4);
  x += __shfl_xor(x, 8);
  return x;
}

// One WMMA with literal nt and literal reuse_a flag (builtin needs ICE modifiers).
#define WMMA_NT(nt, reuse)                                                          \
  {                                                                                 \
    v16h b = *(const v16h*)(Bp + ((size_t)((kt*8 + (nt))*32 + lane) << 4));         \
    acc[nt] = __builtin_amdgcn_wmma_f32_16x16x32_f16(false, a.v, false, b,          \
                                                     (short)0, acc[nt],             \
                                                     reuse, false);                 \
  }
#define WMMA_ROW8()                                                                 \
  WMMA_NT(0, true) WMMA_NT(1, true) WMMA_NT(2, true) WMMA_NT(3, true)               \
  WMMA_NT(4, true) WMMA_NT(5, true) WMMA_NT(6, true) WMMA_NT(7, false)

// ---- WMMA core: one wave computes a 16x128 f32 tile of A[rows,K] @ Bpacked ----
// A fragment per ISA layout: lane L holds row (L&15); K pairs per VGPR group.
template<int KTILES>
__device__ __forceinline__ void gemm_core(const _Float16* A, size_t row, int lane,
                                          const _Float16* Bp, v8f* acc) {
  const int K = KTILES * 32;
  #pragma unroll
  for (int kt = 0; kt < KTILES; ++kt) {
    Frag a;
    #pragma unroll
    for (int g = 0; g < 8; ++g) {
      int kb = kt*32 + ((g >> 2) << 4) + ((lane & 16) ? 8 : 0) + ((g & 3) << 1);
      a.u[g] = *(const unsigned*)(A + row * (size_t)K + kb);
    }
    WMMA_ROW8()
  }
}

// ---- bias + residual + LayerNorm epilogue (row stats across 16 lanes x 8 ntiles) ----
__device__ __forceinline__ void epilogue_ln(const v8f* acc, size_t rowBase, int lane,
                                            const float* bias, const _Float16* Xres,
                                            const float* gam, const float* bet,
                                            _Float16* out) {
  int N = lane & 15;
  int moff = (lane & 16) ? 8 : 0;
  #pragma unroll
  for (int v = 0; v < 8; ++v) {
    size_t row = rowBase + v + moff;
    float vals[8];
    float s1 = 0.f, s2 = 0.f;
    #pragma unroll
    for (int nt = 0; nt < 8; ++nt) {
      int col = nt*16 + N;
      float x = acc[nt][v] + bias[col] + (float)Xres[row*HH + col];
      vals[nt] = x; s1 += x; s2 += x * x;
    }
    s1 = hred16(s1); s2 = hred16(s2);
    float mu  = s1 * (1.f/128.f);
    float var = s2 * (1.f/128.f) - mu*mu;
    float rs  = rsqrtf(var + 1e-5f);
    #pragma unroll
    for (int nt = 0; nt < 8; ++nt) {
      int col = nt*16 + N;
      float y = (vals[nt] - mu) * rs * gam[col] + bet[col];
      out[row*HH + col] = (_Float16)y;
    }
  }
}

// ---- pack f32 weight [Kdim x 128] into WMMA B-fragment lane order (f16), K zero-padded ----
__global__ __launch_bounds__(256) void k_pack(const float* W, _Float16* out, int Kdim, int total) {
  int idx = blockIdx.x * 256 + threadIdx.x;
  if (idx >= total) return;
  int h = idx & 15, lane = (idx >> 4) & 31, tile = idx >> 9;
  int kt = tile >> 3, nt = tile & 7;
  int g = h >> 1, p = h & 1;
  int K = kt*32 + ((g >> 2) << 4) + ((lane & 16) ? 8 : 0) + ((g & 3) << 1) + p;
  int n = nt*16 + (lane & 15);
  out[idx] = (K < Kdim) ? (_Float16)W[(size_t)K * HH + n] : (_Float16)0.f;
}

__device__ __forceinline__ float pe_val(int pos, int col) {
  int i = (col < 64) ? col : (col - 64);
  float ang = (float)pos * __expf(-9.210340371976184f * (float)i / 64.f); // ln(1e4)
  return (col < 64) ? __sinf(ang) : __cosf(ang);
}

// ---- embedding: relu(concat(ef, st) @ Wein + bein) + positional encoding -> f16 ----
__global__ __launch_bounds__(256) void k_embed(const float* ef, const float* st,
                                               const _Float16* WeinP, const float* bein,
                                               _Float16* X0) {
  int lane = threadIdx.x & 31, wave = threadIdx.x >> 5;
  size_t rowBase = (size_t)blockIdx.x * 128 + wave * 16;
  size_t r = rowBase + (lane & 15);
  Frag a;
  #pragma unroll
  for (int g = 0; g < 8; ++g) {
    int kb = ((g >> 2) << 4) + ((lane & 16) ? 8 : 0) + ((g & 3) << 1);
    #pragma unroll
    for (int p = 0; p < 2; ++p) {
      int k = kb + p;
      float x = (k < 16) ? ef[r*16 + k] : ((k == 16) ? st[r] : 0.f);
      a.h[2*g + p] = (_Float16)x;
    }
  }
  v8f acc[8] = {};
  {
    const _Float16* Bp = WeinP;
    const int kt = 0;
    WMMA_ROW8()
  }
  int N = lane & 15, moff = (lane & 16) ? 8 : 0;
  #pragma unroll
  for (int v = 0; v < 8; ++v) {
    size_t row = rowBase + v + moff;
    int pos = (int)(row & 7);
    #pragma unroll
    for (int nt = 0; nt < 8; ++nt) {
      int col = nt*16 + N;
      float x = fmaxf(acc[nt][v] + bein[col], 0.f) + pe_val(pos, col);
      X0[row*HH + col] = (_Float16)x;
    }
  }
}

// ---- fused QKV(WMMA) + 8x8 softmax attention(VALU) + Wo(WMMA) + residual+LN ----
// 64 rows (8 edges) per block, 4 waves. LDS: Q,K,V,O (64KB) + Wo staged async (32KB).
__global__ __launch_bounds__(128) void k_attn_block(const _Float16* Xin,
    const _Float16* WqP, const _Float16* WkP, const _Float16* WvP, const _Float16* WoP,
    const float* bq, const float* bk, const float* bv, const float* bo,
    const float* g1, const float* bt1,
    const int* elen, _Float16* Xout) {
  extern __shared__ _Float16 sm[];
  _Float16* Qs  = sm;             // 64*128 halves each
  _Float16* Ks  = sm + 8192;
  _Float16* Vs  = sm + 16384;
  _Float16* Os  = sm + 24576;
  _Float16* Wol = sm + 32768;     // 16384 halves (32KB) for Wo
  int tid = threadIdx.x;
  int lane = tid & 31, wave = tid >> 5;
  size_t rowBase = (size_t)blockIdx.x * 64 + wave * 16;
  int lrb = wave * 16;
  int N = lane & 15, moff = (lane & 16) ? 8 : 0;
  size_t rowA = rowBase + (lane & 15);

  // kick off async staging of Wo into LDS (overlaps QKV GEMMs + attention)
  {
    unsigned base = (unsigned)(size_t)(const void*)Wol;
    #pragma unroll
    for (int rnd = 0; rnd < 16; ++rnd) {        // 2048 chunks of 16B
      int idx = rnd * 128 + tid;
      async_copy16(base + idx * 16, WoP + (size_t)idx * 8);
    }
  }

  const _Float16* Wp[3]  = {WqP, WkP, WvP};
  const float*    bp3[3] = {bq, bk, bv};
  _Float16*       dst3[3]= {Qs, Ks, Vs};
  #pragma unroll
  for (int m = 0; m < 3; ++m) {
    v8f acc[8] = {};
    gemm_core<4>(Xin, rowA, lane, Wp[m], acc);
    #pragma unroll
    for (int v = 0; v < 8; ++v) {
      int lr = lrb + v + moff;
      #pragma unroll
      for (int nt = 0; nt < 8; ++nt) {
        int col = nt*16 + N;
        dst3[m][lr*HH + col] = (_Float16)(acc[nt][v] + bp3[m][col]);
      }
    }
  }
  __syncthreads();

  if (tid < 64) {   // one thread per (edge, head)
    int el = tid >> 3, head = tid & 7;
    size_t e = (size_t)blockIdx.x * 8 + el;
    int len = elen[e];
    const _Float16* Qb = Qs + (el*8)*HH + head*16;
    const _Float16* Kb = Ks + (el*8)*HH + head*16;
    const _Float16* Vb = Vs + (el*8)*HH + head*16;
    _Float16*       Ob = Os + (el*8)*HH + head*16;
    for (int q = 0; q < 8; ++q) {
      float sc[8]; float mx = -1e30f;
      for (int k2 = 0; k2 < 8; ++k2) {
        float s;
        if (k2 < len) {
          s = 0.f;
          #pragma unroll
          for (int d = 0; d < 16; ++d)
            s += (float)Qb[q*HH + d] * (float)Kb[k2*HH + d];
          s *= 0.25f;            // 1/sqrt(16)
        } else s = -1e9f;
        sc[k2] = s; mx = fmaxf(mx, s);
      }
      float ssum = 0.f;
      for (int k2 = 0; k2 < 8; ++k2) { sc[k2] = __expf(sc[k2] - mx); ssum += sc[k2]; }
      float inv = 1.f / ssum;
      #pragma unroll
      for (int d = 0; d < 16; ++d) {
        float o = 0.f;
        for (int k2 = 0; k2 < 8; ++k2) o += sc[k2] * (float)Vb[k2*HH + d];
        Ob[q*HH + d] = (_Float16)(o * inv);
      }
    }
  }
  wait_async0();          // Wo staged; publish at the barrier below
  __syncthreads();

  v8f acc[8] = {};
  gemm_core<4>(Os, (size_t)(lrb + (lane & 15)), lane, Wol, acc);
  epilogue_ln(acc, rowBase, lane, bo, Xin, g1, bt1, Xout);
}

// ---- fused FFN: relu(X@W1+b1) in LDS, then F@W2(LDS, async-staged) + residual + LN ----
__global__ __launch_bounds__(256) void k_ffn_block(const _Float16* X,
    const _Float16* W1P, const float* b1,
    const _Float16* W2P, const float* b2,
    const float* g2, const float* bt2, _Float16* Xout) {
  extern __shared__ _Float16 sm[];   // Fs 32KB + W2 32KB
  _Float16* Fs  = sm;
  _Float16* W2l = sm + 16384;
  int lane = threadIdx.x & 31, wave = threadIdx.x >> 5;
  size_t rowBase = (size_t)blockIdx.x * 128 + wave * 16;
  int lrb = wave * 16;
  int N = lane & 15, moff = (lane & 16) ? 8 : 0;

  // async stage W2 into LDS; overlaps the W1 GEMM below
  {
    unsigned base = (unsigned)(size_t)(const void*)W2l;
    #pragma unroll
    for (int rnd = 0; rnd < 8; ++rnd) {          // 2048 chunks of 16B
      int idx = rnd * 256 + threadIdx.x;
      async_copy16(base + idx * 16, W2P + (size_t)idx * 8);
    }
  }
  {
    v8f acc[8] = {};
    gemm_core<4>(X, rowBase + (lane & 15), lane, W1P, acc);
    #pragma unroll
    for (int v = 0; v < 8; ++v) {
      int lr = lrb + v + moff;
      #pragma unroll
      for (int nt = 0; nt < 8; ++nt) {
        int col = nt*16 + N;
        Fs[lr*HH + col] = (_Float16)fmaxf(acc[nt][v] + b1[col], 0.f);
      }
    }
  }
  wait_async0();
  __syncthreads();
  v8f acc[8] = {};
  gemm_core<4>(Fs, (size_t)(lrb + (lane & 15)), lane, W2l, acc);
  epilogue_ln(acc, rowBase, lane, b2, X, g2, bt2, Xout);
}

// ---- generic GEMM + bias + relu (f16 or f32 out) for the K=256 concat projections ----
template<int KTILES, int F32OUT>
__global__ __launch_bounds__(256) void k_gemm_relu(const _Float16* A, const _Float16* Bp,
                                                   const float* bias, _Float16* outh, float* outf) {
  int lane = threadIdx.x & 31, wave = threadIdx.x >> 5;
  size_t rowBase = (size_t)blockIdx.x * 128 + wave * 16;
  v8f acc[8] = {};
  gemm_core<KTILES>(A, rowBase + (lane & 15), lane, Bp, acc);
  int N = lane & 15, moff = (lane & 16) ? 8 : 0;
  #pragma unroll
  for (int v = 0; v < 8; ++v) {
    size_t row = rowBase + v + moff;
    #pragma unroll
    for (int nt = 0; nt < 8; ++nt) {
      int col = nt*16 + N;
      float x = fmaxf(acc[nt][v] + bias[col], 0.f);
      if (F32OUT) outf[row*HH + col] = x;
      else        outh[row*HH + col] = (_Float16)x;
    }
  }
}

// ---- tail kernels ----
__global__ __launch_bounds__(256) void k_gather(const _Float16* Xa, const int* elen, _Float16* eo) {
  size_t idx = (size_t)blockIdx.x * 256 + threadIdx.x;
  if (idx >= (size_t)EE * HH) return;
  size_t e = idx >> 7; int c = idx & 127;
  int s = elen[e] - 1; s = s < 0 ? 0 : (s > 7 ? 7 : s);
  eo[idx] = Xa[(e*SS + s)*HH + c];
}

__global__ __launch_bounds__(256) void k_aval(const _Float16* Xa, const int* elen,
                                              const float* Wat, float* aval) {
  size_t e = (size_t)blockIdx.x * 256 + threadIdx.x;
  if (e >= EE) return;
  int s = elen[e] - 1; s = s < 0 ? 0 : (s > 7 ? 7 : s);
  const _Float16* xr = Xa + (e*SS + s)*HH;
  float acc = 0.f;
  for (int c = 0; c < HH; ++c) acc += (float)xr[c] * Wat[c];
  aval[e] = (acc > 0.f) ? acc : 0.01f * acc;
}

__global__ __launch_bounds__(128) void k_sparsemax(const float* aval, float* alpha) {
  int d = blockIdx.x * 128 + threadIdx.x;
  if (d >= NDST) return;
  float z[16], zs[16];
  float mx = -1e30f;
  for (int j = 0; j < 16; ++j) { z[j] = aval[d*16 + j]; mx = fmaxf(mx, z[j]); }
  for (int j = 0; j < 16; ++j) { z[j] -= mx; zs[j] = z[j]; }
  for (int i = 1; i < 16; ++i) {          // insertion sort descending
    float key = zs[i]; int j = i - 1;
    while (j >= 0 && zs[j] < key) { zs[j+1] = zs[j]; --j; }
    zs[j+1] = key;
  }
  float cs = 0.f, ssum = 0.f, kk = 1.f;
  for (int r = 1; r <= 16; ++r) {
    float zr = zs[r-1];
    cs += zr;
    if (1.f + (float)r * zr > cs) { kk = (float)r; ssum += zr; }
  }
  float tau = (ssum - 1.f) / kk;
  for (int j = 0; j < 16; ++j) alpha[d*16 + j] = fmaxf(0.f, z[j] - tau);
}

__global__ __launch_bounds__(256) void k_build_am(const float* nodef, const int* srci,
                                                  const _Float16* eo, _Float16* Am) {
  size_t idx = (size_t)blockIdx.x * 256 + threadIdx.x;
  if (idx >= (size_t)EE * 256) return;
  size_t e = idx >> 8; int c = idx & 255;
  Am[idx] = (c < HH) ? (_Float16)nodef[(size_t)srci[e]*HH + c]
                     : eo[e*HH + (c - HH)];
}

__global__ __launch_bounds__(256) void k_agg(const float* nodef, const _Float16* mbuf,
                                             const float* alpha, _Float16* Af) {
  size_t idx = (size_t)blockIdx.x * 256 + threadIdx.x;
  if (idx >= (size_t)NDST * HH) return;
  size_t d = idx >> 7; int c = idx & 127;
  float s = 0.f;
  #pragma unroll
  for (int j = 0; j < 16; ++j)
    s += alpha[d*16 + j] * (float)mbuf[(d*16 + j)*HH + c];
  Af[d*256 + c]      = (_Float16)nodef[d*HH + c];  // h_dst
  Af[d*256 + HH + c] = (_Float16)s;                // h_neigh
}

extern "C" void kernel_launch(void* const* d_in, const int* in_sizes, int n_in,
                              void* d_out, int out_size, void* d_ws, size_t ws_size,
                              hipStream_t stream) {
  (void)in_sizes; (void)n_in; (void)out_size; (void)ws_size;
  // setup_inputs() insertion order, params flattened in insertion order.
  const float* nodef = (const float*)d_in[0];
  const float* ef    = (const float*)d_in[1];
  const float* st    = (const float*)d_in[2];
  const int*   elen  = (const int*)d_in[4];
  const int*   srci  = (const int*)d_in[5];
  const float* Wein  = (const float*)d_in[6];
  const float* bein  = (const float*)d_in[7];
  const float* Wattn = (const float*)d_in[72];
  const float* Weout = (const float*)d_in[73];
  const float* beout = (const float*)d_in[74];
  const float* Wnode = (const float*)d_in[75];
  const float* bnode = (const float*)d_in[76];

  char* ws = (char*)d_ws;
  size_t off = 0;
  auto alloc = [&](size_t bytes) -> void* {
    off = (off + 255) & ~(size_t)255;
    void* p = ws + off; off += bytes; return p;
  };
  _Float16* X0    = (_Float16*)alloc((size_t)ROWS_ES * HH * 2);
  _Float16* Xa    = (_Float16*)alloc((size_t)ROWS_ES * HH * 2);
  _Float16* eo    = (_Float16*)alloc((size_t)EE * HH * 2);
  _Float16* Am    = (_Float16*)alloc((size_t)EE * 256 * 2);
  _Float16* mbuf  = (_Float16*)alloc((size_t)EE * HH * 2);
  float*    aval  = (float*)alloc((size_t)EE * 4);
  float*    alpha = (float*)alloc((size_t)EE * 4);
  _Float16* Af    = (_Float16*)alloc((size_t)NDST * 256 * 2);
  _Float16* WP[4][6];
  for (int L = 0; L < 4; ++L)
    for (int m = 0; m < 6; ++m)
      WP[L][m] = (_Float16*)alloc(32768);
  _Float16* WeinP  = (_Float16*)alloc(8192);
  _Float16* WeoutP = (_Float16*)alloc(65536);
  _Float16* WnodeP = (_Float16*)alloc(65536);

  auto pack = [&](const float* W, _Float16* dst, int Kdim, int ktiles) {
    int total = ktiles * 4096;
    k_pack<<<(total + 255) / 256, 256, 0, stream>>>(W, dst, Kdim, total);
  };
  static const int widx[6] = {0, 2, 4, 6, 10, 12};  // Wq,Wk,Wv,Wo,W1,W2 within layer dict
  for (int L = 0; L < 4; ++L) {
    int bi = 8 + L * 16;
    for (int m = 0; m < 6; ++m)
      pack((const float*)d_in[bi + widx[m]], WP[L][m], 128, 4);
  }
  pack(Wein,  WeinP, 17, 1);
  pack(Weout, WeoutP, 256, 8);
  pack(Wnode, WnodeP, 256, 8);

  k_embed<<<ROWS_ES / 128, 256, 0, stream>>>(ef, st, WeinP, bein, X0);

  for (int t = 0; t < 2; ++t) {       // t1 then t2, both starting from X0
    for (int l = 0; l < 2; ++l) {
      int L = t * 2 + l;
      int bi = 8 + L * 16;
      const _Float16* Xin = (l == 0) ? X0 : Xa;
      k_attn_block<<<ROWS_ES / 64, 128, (4*8192 + 16384) * 2, stream>>>(Xin,
          WP[L][0], WP[L][1], WP[L][2], WP[L][3],
          (const float*)d_in[bi+1], (const float*)d_in[bi+3],
          (const float*)d_in[bi+5], (const float*)d_in[bi+7],
          (const float*)d_in[bi+8], (const float*)d_in[bi+9],
          elen, Xa);
      k_ffn_block<<<ROWS_ES / 128, 256, (16384 + 16384) * 2, stream>>>(Xa,
          WP[L][4], (const float*)d_in[bi+11],
          WP[L][5], (const float*)d_in[bi+13],
          (const float*)d_in[bi+14], (const float*)d_in[bi+15], Xa);
    }
    if (t == 0)
      k_gather<<<(EE * HH + 255) / 256, 256, 0, stream>>>(Xa, elen, eo);
    else
      k_aval<<<(EE + 255) / 256, 256, 0, stream>>>(Xa, elen, Wattn, aval);
  }

  k_sparsemax<<<(NDST + 127) / 128, 128, 0, stream>>>(aval, alpha);
  k_build_am<<<(EE * 256 + 255) / 256, 256, 0, stream>>>(nodef, srci, eo, Am);
  k_gemm_relu<8, 0><<<EE / 128, 256, 0, stream>>>(Am, WeoutP, beout, mbuf, nullptr);
  k_agg<<<(NDST * HH + 255) / 256, 256, 0, stream>>>(nodef, mbuf, alpha, Af);
  k_gemm_relu<8, 1><<<NDST / 128, 256, 0, stream>>>(Af, WnodeP, bnode, nullptr, (float*)d_out);
}
